// BinaryLeNet5_71141838291230
// MI455X (gfx1250) — matrix operand order repaired and checked by reading, not compiled, not run on hardware
//
#include <hip/hip_runtime.h>
#include <hip/hip_bf16.h>

// ---------------------------------------------------------------------------
// BinaryLeNet5 forward, fused pipeline for gfx1250 (MI455X), wave32 + WMMA.
//   conv1 (scalar, C_in=1) -> relu -> pool
//   conv2 as im2col GEMM via v_wmma_f32_16x16x32_f16 (M=144,N=16,K=150->160)
//   fc1/fc2/fc3 as GEMMs via v_wmma_f32_16x16x32_f16
//   A operands staged in LDS in WMMA-fragment layout -> contiguous ds_load_b128
//   per-row means + global betas via deterministic tree reductions (no atomics)
//   log_softmax epilogue
// ---------------------------------------------------------------------------

typedef __attribute__((ext_vector_type(16))) _Float16 v16h;
typedef __attribute__((ext_vector_type(8)))  float    v8f;

union FV { v8f v; float e[8]; };

__device__ __forceinline__ float signf(float v) {
    return v > 0.f ? 1.f : (v < 0.f ? -1.f : 0.f);
}

// Inverse A-fragment map (16-bit A 16x32, ISA 7.12.2): given k-in-tile kl
// (0..31), which lane group and which half-slot holds it.
//   group0 holds kl in {0..7, 16..23}; group1 holds {8..15, 24..31}
//   h = (kl&7) | ((kl&16)>>1)
__device__ __forceinline__ int fragGrp(int kl) { return (kl >> 3) & 1; }
__device__ __forceinline__ int fragH(int kl)   { return (kl & 7) | ((kl & 16) >> 1); }

// Forward map: k-in-tile held at half slot h by lane group grp.
__device__ __forceinline__ int kmapA(int grp, int h) {
    int r = h >> 1, p = h & 1;
    int base = (r < 4) ? (r * 2 + p) : (r * 2 + 8 + p);
    return base + grp * 8;
}

// ---------------------------------------------------------------------------
// Per-row mean + per-row sum|x-mean|
// ---------------------------------------------------------------------------
__global__ void k_rowstats(const float* __restrict__ in, float* __restrict__ rowmean,
                           float* __restrict__ rowabs, int rowlen) {
    int row = blockIdx.x;
    const float* p = in + (size_t)row * rowlen;
    __shared__ float red[256];
    float s = 0.f;
    for (int i = threadIdx.x; i < rowlen; i += 256) s += p[i];
    red[threadIdx.x] = s;
    __syncthreads();
    for (int o = 128; o > 0; o >>= 1) {
        if (threadIdx.x < o) red[threadIdx.x] += red[threadIdx.x + o];
        __syncthreads();
    }
    float mean = red[0] / (float)rowlen;
    if (threadIdx.x == 0) rowmean[row] = mean;
    __syncthreads();
    float a = 0.f;
    for (int i = threadIdx.x; i < rowlen; i += 256) a += fabsf(p[i] - mean);
    red[threadIdx.x] = a;
    __syncthreads();
    for (int o = 128; o > 0; o >>= 1) {
        if (threadIdx.x < o) red[threadIdx.x] += red[threadIdx.x + o];
        __syncthreads();
    }
    if (threadIdx.x == 0) rowabs[row] = red[0];
}

// beta = (sum of rowabs) * inv_total   (single block, deterministic)
__global__ void k_beta(const float* __restrict__ rowabs, int nrows, float inv_total,
                       float* __restrict__ beta_out) {
    __shared__ float red[256];
    float s = 0.f;
    for (int i = threadIdx.x; i < nrows; i += 256) s += rowabs[i];
    red[threadIdx.x] = s;
    __syncthreads();
    for (int o = 128; o > 0; o >>= 1) {
        if (threadIdx.x < o) red[threadIdx.x] += red[threadIdx.x + o];
        __syncthreads();
    }
    if (threadIdx.x == 0) beta_out[0] = red[0] * inv_total;
}

// ---------------------------------------------------------------------------
// Pack sign(W) into WMMA B-fragment layout: [nt][kt][lane][16 halfs].
// B (32x16, f16): lane group 0 -> K=0..7,16..23, group 1 -> K=8..15,24..31 (table order).
// ---------------------------------------------------------------------------
__global__ void k_packB(const float* __restrict__ w, int O, int K, int Kt,
                        _Float16* __restrict__ out) {
    int blk = blockIdx.x;           // blk = nt*Kt + kt
    int lane = threadIdx.x;         // 32 threads
    int kt = blk % Kt;
    int nt = blk / Kt;
    int n = nt * 16 + (lane & 15);
    int grp = lane >> 4;
    _Float16* dst = out + ((size_t)blk * 32 + lane) * 16;
    for (int h = 0; h < 16; ++h) {
        int k = kt * 32 + (h & 7) + 8 * grp + ((h >> 3) << 4);
        float v = (n < O && k < K) ? signf(w[(size_t)n * K + k]) : 0.f;
        dst[h] = (_Float16)v;
    }
}

// ---------------------------------------------------------------------------
// conv1: [B,1,32,32] -> sign conv 5x5 pad2 -> *a1*beta1 -> relu -> pool2
// ---------------------------------------------------------------------------
__global__ void k_conv1(const float* __restrict__ x, const float* __restrict__ w1,
                        const float* __restrict__ a1, const float* __restrict__ beta1,
                        const float* __restrict__ mean1, float* __restrict__ h1) {
    int b = blockIdx.x;
    __shared__ float xs[1024];
    __shared__ float wsg[150];
    float mean = mean1[b];
    const float* xp = x + (size_t)b * 1024;
    for (int i = threadIdx.x; i < 1024; i += 256) xs[i] = signf(xp[i] - mean);
    if (threadIdx.x < 150) wsg[threadIdx.x] = signf(w1[threadIdx.x]);
    __syncthreads();
    float scale = a1[0] * beta1[0];
    for (int po = threadIdx.x; po < 1536; po += 256) {
        int c = po >> 8;
        int py = (po >> 4) & 15;
        int px = po & 15;
        float best = 0.f;  // relu floor
        for (int dy = 0; dy < 2; ++dy)
        for (int dx = 0; dx < 2; ++dx) {
            int oy = py * 2 + dy, ox = px * 2 + dx;
            float s = 0.f;
            for (int ky = 0; ky < 5; ++ky) {
                int iy = oy + ky - 2;
                if ((unsigned)iy >= 32u) continue;
                for (int kx = 0; kx < 5; ++kx) {
                    int ix = ox + kx - 2;
                    if ((unsigned)ix >= 32u) continue;
                    s += xs[iy * 32 + ix] * wsg[c * 25 + ky * 5 + kx];
                }
            }
            float v = s * scale;
            if (v > best) best = v;
        }
        h1[(size_t)b * 1536 + po] = best;
    }
}

// ---------------------------------------------------------------------------
// conv2 via WMMA: one wave per sample.
//  Pass 1: signs -> LDS (7 channels; channel 6 zeroed = zero-pad for K>=150)
//  Pass 2: cooperative im2col into LDS in A-fragment layout [mt][kt][lane][16]
//  Pass 3: 9x5 WMMA with contiguous LDS fragment loads
//  Pass 4: *a2*beta2 -> relu -> pool2 -> h2 [B,576] (c*36 + py*6 + px)
// ---------------------------------------------------------------------------
__global__ void __launch_bounds__(32)
k_conv2(const float* __restrict__ h1, const float* __restrict__ mean2,
        const float* __restrict__ a2, const float* __restrict__ beta2,
        const _Float16* __restrict__ Bp, float* __restrict__ h2) {
    int b = blockIdx.x;
    int lane = threadIdx.x;
    __shared__ _Float16 sgn[7 * 256];            // 6ch x 16x16 + zero channel
    __shared__ _Float16 afrag[9 * 5 * 32 * 16];  // A fragments [mt][kt][lane][h]
    __shared__ float dres[144 * 16];             // conv out [m=oy*12+ox][c]

    const float* hp = h1 + (size_t)b * 1536;
    for (int i = lane; i < 7 * 256; i += 32) {
        float v = 0.f;
        if (i < 1536) v = signf(hp[i] - mean2[b * 6 + (i >> 8)]);
        sgn[i] = (_Float16)v;
    }
    __syncthreads();

    // Build A fragments: slot i = ((mt*5 + kt)*32 + laneF)*16 + h
    for (int i = lane; i < 9 * 5 * 32 * 16; i += 32) {
        int h = i & 15;
        int laneF = (i >> 4) & 31;
        int ktmt = i >> 9;
        int kt = ktmt % 5;
        int mt = ktmt / 5;
        int m = mt * 16 + (laneF & 15);
        int grp = laneF >> 4;
        int k = kt * 32 + kmapA(grp, h);   // 0..159; >=150 lands in zero channel
        int oy = m / 12, ox = m - oy * 12;
        int c = k / 25, t = k - c * 25;    // c==6 for padded k
        int ky = t / 5, kx = t - ky * 5;
        afrag[i] = sgn[c * 256 + (oy + ky) * 16 + (ox + kx)];
    }
    __syncthreads();

    int ml = lane & 15, grp = lane >> 4;
    for (int mt = 0; mt < 9; ++mt) {
        FV acc;
        for (int r = 0; r < 8; ++r) acc.e[r] = 0.f;
        for (int kt = 0; kt < 5; ++kt) {
            v16h av = *(const v16h*)(afrag + ((size_t)(mt * 5 + kt) * 32 + lane) * 16);
            v16h bv = *(const v16h*)(Bp + ((size_t)kt * 32 + lane) * 16);
            acc.v = __builtin_amdgcn_wmma_f32_16x16x32_f16(
                false, av, false, bv, (short)0, acc.v, false, false);
        }
        for (int r = 0; r < 8; ++r)
            dres[(mt * 16 + r + 8 * grp) * 16 + ml] = acc.e[r];
    }
    __syncthreads();

    float scale = a2[0] * beta2[0];
    for (int i = lane; i < 576; i += 32) {
        int c = i / 36;
        int py = (i % 36) / 6;
        int px = i % 6;
        float best = 0.f;  // relu floor
        for (int dy = 0; dy < 2; ++dy)
        for (int dx = 0; dx < 2; ++dx) {
            int oy = py * 2 + dy, ox = px * 2 + dx;
            float v = dres[(oy * 12 + ox) * 16 + c] * scale;
            if (v > best) best = v;
        }
        h2[(size_t)b * 576 + i] = best;
    }
}

// ---------------------------------------------------------------------------
// Generic binary FC layer via WMMA: one wave per 16-row M-tile, all N-tiles.
// A staged in LDS in fragment layout [kt][lane][16] -> contiguous loads.
// ---------------------------------------------------------------------------
__global__ void __launch_bounds__(32)
k_fc(const float* __restrict__ in, const float* __restrict__ rowmean,
     const float* __restrict__ alpha, const float* __restrict__ beta,
     const _Float16* __restrict__ Bp, float* __restrict__ out,
     int K, int Kt, int O, int Nt, int doRelu) {
    extern __shared__ _Float16 As[];  // Kt*32*16 halfs, fragment layout
    int b0 = blockIdx.x * 16;
    int lane = threadIdx.x;
    int Kpad = Kt * 32;
    // Stage signs into fragment layout: (row m, col k) -> slot.
    for (int i = lane; i < 16 * Kpad; i += 32) {
        int row = i / Kpad, col = i - row * Kpad;
        float v = 0.f;
        if (col < K) v = signf(in[(size_t)(b0 + row) * K + col] - rowmean[b0 + row]);
        int kt = col >> 5, kl = col & 31;
        int slot = ((kt * 32 + fragGrp(kl) * 16 + row) << 4) | fragH(kl);
        As[slot] = (_Float16)v;
    }
    __syncthreads();
    float scale = alpha[0] * beta[0];
    int ml = lane & 15, grp = lane >> 4;
    for (int nt = 0; nt < Nt; ++nt) {
        __builtin_prefetch(Bp + ((size_t)(nt * Kt) * 32) * 16, 0, 1);
        FV acc;
        for (int r = 0; r < 8; ++r) acc.e[r] = 0.f;
        for (int kt = 0; kt < Kt; ++kt) {
            v16h av = *(const v16h*)(As + ((size_t)kt * 32 + lane) * 16);
            v16h bv = *(const v16h*)(Bp + ((size_t)(nt * Kt + kt) * 32 + lane) * 16);
            acc.v = __builtin_amdgcn_wmma_f32_16x16x32_f16(
                false, av, false, bv, (short)0, acc.v, false, false);
        }
        int n = nt * 16 + ml;
        if (n < O) {
            for (int r = 0; r < 8; ++r) {
                int m = b0 + r + 8 * grp;
                float v = acc.e[r] * scale;
                if (doRelu && v < 0.f) v = 0.f;
                out[(size_t)m * O + n] = v;
            }
        }
    }
}

// ---------------------------------------------------------------------------
// log_softmax over the 10 logits of each sample
// ---------------------------------------------------------------------------
__global__ void k_lsm(const float* __restrict__ h5, float* __restrict__ out, int Bn) {
    int i = blockIdx.x * blockDim.x + threadIdx.x;
    if (i >= Bn) return;
    const float* p = h5 + (size_t)i * 10;
    float m = p[0];
    for (int j = 1; j < 10; ++j) m = fmaxf(m, p[j]);
    float s = 0.f;
    for (int j = 0; j < 10; ++j) s += __expf(p[j] - m);
    float lse = __logf(s) + m;
    for (int j = 0; j < 10; ++j) out[(size_t)i * 10 + j] = p[j] - lse;
}

// ---------------------------------------------------------------------------
extern "C" void kernel_launch(void* const* d_in, const int* in_sizes, int n_in,
                              void* d_out, int out_size, void* d_ws, size_t ws_size,
                              hipStream_t stream) {
    const float* x   = (const float*)d_in[0];
    const float* w1  = (const float*)d_in[1];
    const float* a1  = (const float*)d_in[2];
    const float* w2  = (const float*)d_in[3];
    const float* a2  = (const float*)d_in[4];
    const float* wf1 = (const float*)d_in[5];
    const float* af1 = (const float*)d_in[6];
    const float* wf2 = (const float*)d_in[7];
    const float* af2 = (const float*)d_in[8];
    const float* wf3 = (const float*)d_in[9];
    const float* af3 = (const float*)d_in[10];

    const int Bn = in_sizes[0] / 1024;  // 8192

    // Workspace carve-up (float units, 256B-aligned slices).
    float* W = (float*)d_ws;
    size_t o = 0;
    auto alloc = [&](size_t n) { size_t r = o; o += (n + 63) & ~(size_t)63; return r; };
    size_t betas  = alloc(8);                       // beta1..beta5
    size_t mean1  = alloc((size_t)Bn);
    size_t mean2  = alloc((size_t)Bn * 6);
    size_t mean3  = alloc((size_t)Bn);
    size_t mean4  = alloc((size_t)Bn);
    size_t mean5  = alloc((size_t)Bn);
    size_t rowabs = alloc((size_t)Bn * 6);
    size_t h1     = alloc((size_t)Bn * 1536);
    size_t h2     = alloc((size_t)Bn * 576);
    size_t h3     = alloc((size_t)Bn * 120);
    size_t h4     = alloc((size_t)Bn * 84);
    size_t h5     = alloc((size_t)Bn * 10);
    size_t bp2    = alloc(1 * 5 * 32 * 16 / 2);     // conv2 packed B (halfs/2)
    size_t bpf1   = alloc(8 * 18 * 32 * 16 / 2);
    size_t bpf2   = alloc(6 * 4 * 32 * 16 / 2);
    size_t bpf3   = alloc(1 * 3 * 32 * 16 / 2);
    (void)ws_size; (void)n_in; (void)out_size;

    _Float16* Bp2  = (_Float16*)(W + bp2);
    _Float16* Bpf1 = (_Float16*)(W + bpf1);
    _Float16* Bpf2 = (_Float16*)(W + bpf2);
    _Float16* Bpf3 = (_Float16*)(W + bpf3);

    // Pack binary weights into WMMA B-fragment layout (activation-independent).
    k_packB<<<1 * 5, 32, 0, stream>>>(w2, 16, 150, 5, Bp2);
    k_packB<<<8 * 18, 32, 0, stream>>>(wf1, 120, 576, 18, Bpf1);
    k_packB<<<6 * 4, 32, 0, stream>>>(wf2, 84, 120, 4, Bpf2);
    k_packB<<<1 * 3, 32, 0, stream>>>(wf3, 10, 84, 3, Bpf3);

    // Stage 1: conv1
    k_rowstats<<<Bn, 256, 0, stream>>>(x, W + mean1, W + rowabs, 1024);
    k_beta<<<1, 256, 0, stream>>>(W + rowabs, Bn, 1.f / ((float)Bn * 1024.f), W + betas + 0);
    k_conv1<<<Bn, 256, 0, stream>>>(x, w1, a1, W + betas + 0, W + mean1, W + h1);

    // Stage 2: conv2 (WMMA)
    k_rowstats<<<Bn * 6, 256, 0, stream>>>(W + h1, W + mean2, W + rowabs, 256);
    k_beta<<<1, 256, 0, stream>>>(W + rowabs, Bn * 6, 1.f / ((float)Bn * 1536.f), W + betas + 1);
    k_conv2<<<Bn, 32, 0, stream>>>(W + h1, W + mean2, a2, W + betas + 1, Bp2, W + h2);

    // Stage 3: fc1 (WMMA)
    k_rowstats<<<Bn, 256, 0, stream>>>(W + h2, W + mean3, W + rowabs, 576);
    k_beta<<<1, 256, 0, stream>>>(W + rowabs, Bn, 1.f / ((float)Bn * 576.f), W + betas + 2);
    k_fc<<<Bn / 16, 32, 18 * 32 * 16 * 2, stream>>>(W + h2, W + mean3, af1, W + betas + 2,
                                                    Bpf1, W + h3, 576, 18, 120, 8, 1);

    // Stage 4: fc2 (WMMA)
    k_rowstats<<<Bn, 256, 0, stream>>>(W + h3, W + mean4, W + rowabs, 120);
    k_beta<<<1, 256, 0, stream>>>(W + rowabs, Bn, 1.f / ((float)Bn * 120.f), W + betas + 3);
    k_fc<<<Bn / 16, 32, 4 * 32 * 16 * 2, stream>>>(W + h3, W + mean4, af2, W + betas + 3,
                                                   Bpf2, W + h4, 120, 4, 84, 6, 1);

    // Stage 5: fc3 (WMMA) + log_softmax
    k_rowstats<<<Bn, 256, 0, stream>>>(W + h4, W + mean5, W + rowabs, 84);
    k_beta<<<1, 256, 0, stream>>>(W + rowabs, Bn, 1.f / ((float)Bn * 84.f), W + betas + 4);
    k_fc<<<Bn / 16, 32, 3 * 32 * 16 * 2, stream>>>(W + h4, W + mean5, af3, W + betas + 4,
                                                   Bpf3, W + h5, 84, 3, 10, 1, 0);
    k_lsm<<<(Bn + 255) / 256, 256, 0, stream>>>(W + h5, (float*)d_out, Bn);
}
